// PerceiverResamplerModule_85813446574265
// MI455X (gfx1250) — matrix operand order repaired and checked by reading, not compiled, not run on hardware
//
#include <hip/hip_runtime.h>
#include <cstdint>
#include <cstddef>

// ---------------- problem constants (from reference) ----------------
#define TT    4
#define FRN   256
#define VV    64
#define DD    1024
#define SS    (FRN * VV)      // 16384 visual tokens
#define NLAT  721
#define INNER 128             // heads * dim_head
#define NHEAD 2
#define DH    64
#define FFDIM 4096
#define LTOT  (SS + NLAT)     // 17105 attention columns
#define LPAD  17120           // LTOT rounded up to 32

typedef __attribute__((ext_vector_type(16))) _Float16 v16h;
typedef __attribute__((ext_vector_type(8)))  _Float16 v8h;
typedef __attribute__((ext_vector_type(8)))  float    v8f;

__device__ __forceinline__ v16h cat8(v8h lo, v8h hi) {
  return __builtin_shufflevector(lo, hi, 0,1,2,3,4,5,6,7,8,9,10,11,12,13,14,15);
}

// ---------------- f32 -> f16 convert with transpose: src K x N  ->  dst N x K --------
__global__ void k_cvt_t(const float* __restrict__ src, _Float16* __restrict__ dst,
                        int K, int N) {
  size_t i = (size_t)blockIdx.x * 256 + threadIdx.x;
  size_t n = (size_t)K * N;
  if (i >= n) return;
  int k  = (int)(i / N);
  int nn = (int)(i % N);
  dst[(size_t)nn * K + k] = (_Float16)src[i];
}

// ---------------- latents init: broadcast + img_cls on row 0 ----------------
__global__ void k_init_latents(const float* __restrict__ lat,
                               const float* __restrict__ cls,
                               float* __restrict__ out) {
  size_t i = (size_t)blockIdx.x * 256 + threadIdx.x;
  const size_t n = (size_t)TT * NLAT * DD;
  if (i >= n) return;
  int t   = (int)(i / ((size_t)NLAT * DD));
  int rem = (int)(i % ((size_t)NLAT * DD));
  int li  = rem / DD, d = rem % DD;
  float v = lat[rem];
  if (li == 0) v += cls[t * DD + d];
  out[i] = v;
}

// ---------------- LayerNorm over D=1024, generic rows -> f16/f32 ----------------
__global__ void k_ln_rows(const float* __restrict__ X, const float* __restrict__ g,
                          const float* __restrict__ b, void* __restrict__ out,
                          int out_f16) {
  __shared__ float red[256];
  const int row = blockIdx.x;
  const int tid = threadIdx.x;
  const float* xp = X + (size_t)row * DD;
  float vals[4]; float sum = 0.f;
#pragma unroll
  for (int i = 0; i < 4; ++i) { int d = tid + i * 256; vals[i] = xp[d]; sum += vals[i]; }
  red[tid] = sum; __syncthreads();
  for (int st = 128; st > 0; st >>= 1) { if (tid < st) red[tid] += red[tid + st]; __syncthreads(); }
  const float mean = red[0] * (1.0f / DD);
  __syncthreads();
  float sq = 0.f;
#pragma unroll
  for (int i = 0; i < 4; ++i) { float dv = vals[i] - mean; sq += dv * dv; }
  red[tid] = sq; __syncthreads();
  for (int st = 128; st > 0; st >>= 1) { if (tid < st) red[tid] += red[tid + st]; __syncthreads(); }
  const float rstd = rsqrtf(red[0] * (1.0f / DD) + 1e-5f);
#pragma unroll
  for (int i = 0; i < 4; ++i) {
    int d = tid + i * 256;
    float y = (vals[i] - mean) * rstd * g[d] + b[d];
    if (out_f16) ((_Float16*)out)[(size_t)row * DD + d] = (_Float16)y;
    else         ((float*)out)[(size_t)row * DD + d]    = y;
  }
}

// ---------------- LN of (x + spatial_emb + frame_emb) -> f16 ----------------
__global__ void k_ln_x_embed(const float* __restrict__ x, const float* __restrict__ sp,
                             const float* __restrict__ fr, const float* __restrict__ g,
                             const float* __restrict__ b, _Float16* __restrict__ out) {
  __shared__ float red[256];
  const int row = blockIdx.x;            // 0 .. T*S-1
  const int s = row % SS;
  const int f = s / VV, v = s % VV;
  const int tid = threadIdx.x;
  const float* xp  = x  + (size_t)row * DD;
  const float* spp = sp + (size_t)v * DD;
  const float* frp = fr + (size_t)f * DD;
  float vals[4]; float sum = 0.f;
#pragma unroll
  for (int i = 0; i < 4; ++i) {
    int d = tid + i * 256;
    float val = xp[d] + spp[d] + frp[d];
    vals[i] = val; sum += val;
  }
  red[tid] = sum; __syncthreads();
  for (int st = 128; st > 0; st >>= 1) { if (tid < st) red[tid] += red[tid + st]; __syncthreads(); }
  const float mean = red[0] * (1.0f / DD);
  __syncthreads();
  float sq = 0.f;
#pragma unroll
  for (int i = 0; i < 4; ++i) { float dv = vals[i] - mean; sq += dv * dv; }
  red[tid] = sq; __syncthreads();
  for (int st = 128; st > 0; st >>= 1) { if (tid < st) red[tid] += red[tid + st]; __syncthreads(); }
  const float rstd = rsqrtf(red[0] * (1.0f / DD) + 1e-5f);
#pragma unroll
  for (int i = 0; i < 4; ++i) {
    int d = tid + i * 256;
    out[(size_t)row * DD + d] = (_Float16)((vals[i] - mean) * rstd * g[d] + b[d]);
  }
}

// ------- WMMA GEMM: C(MxN) = A(MxK,f16) * B(KxN) with B pre-transposed (Bt: N x K) ---
// One wave per block computes a 16x32 C macro-tile (2 N-tiles sharing one A fragment).
// Out-of-range A rows are clamped (their C rows are never stored), so the inner loop
// has no conditionals: pure b128 loads + 2x v_wmma per 32-K step.
// flags bit0: store f16 ; bit1: exact GELU epilogue ; resid: optional f32 add
__global__ void k_gemm(const _Float16* __restrict__ A, const _Float16* __restrict__ Bt,
                       const float* __restrict__ resid, void* __restrict__ Cout,
                       int M, int N, int K, int flags) {
  const int lane = threadIdx.x & 31;
  const int l15  = lane & 15;
  const int hi   = lane >> 4;
  const int tn = blockIdx.x, tm = blockIdx.y;
  int row = tm * 16 + l15;                    // A row handled by this lane
  if (row >= M) row = M - 1;                  // clamp: affects only unstored C rows
  const int col0 = tn * 32 + l15;             // first C column handled by this lane
  const int kbase = hi * 16;                  // A-fragment K base (contiguous layout)
  const int kn    = hi * 16;                  // B-fragment K base
  const _Float16* ap  = A  + (size_t)row * K + kbase;
  const _Float16* bp0 = Bt + (size_t)col0 * K + kn;        // contiguous 16-half frags
  const _Float16* bp1 = bp0 + (size_t)16 * K;
  v8f acc0 = {}, acc1 = {};
  for (int k0 = 0; k0 < K; k0 += 32) {
    v16h a  = cat8(*(const v8h*)(ap + k0), *(const v8h*)(ap + k0 + 8));
    v16h b0 = *(const v16h*)(bp0 + k0);
    v16h b1 = *(const v16h*)(bp1 + k0);
    if (k0 + 32 < K) __builtin_prefetch(ap + k0 + 32, 0, 3);
    acc0 = __builtin_amdgcn_wmma_f32_16x16x32_f16(false, a, false, b0,
                                                  (short)0, acc0, false, false);
    acc1 = __builtin_amdgcn_wmma_f32_16x16x32_f16(false, a, false, b1,
                                                  (short)0, acc1, false, false);
  }
#pragma unroll
  for (int r = 0; r < 8; ++r) {
    int orow = tm * 16 + r + hi * 8;          // C layout: lanes>=16 hold M = r+8
    if (orow < M) {
#pragma unroll
      for (int nt = 0; nt < 2; ++nt) {
        int col = col0 + nt * 16;
        float v = (nt == 0) ? acc0[r] : acc1[r];
        if (resid) v += resid[(size_t)orow * N + col];
        if (flags & 2) v = 0.5f * v * (1.0f + erff(v * 0.70710678118654752f));
        if (flags & 1) ((_Float16*)Cout)[(size_t)orow * N + col] = (_Float16)v;
        else           ((float*)Cout)[(size_t)orow * N + col]    = v;
      }
    }
  }
}

// ------- repack KV into zero-padded attention layouts ----------
//   Kc[(t*LPAD + c)*128 + i]                  (token-major keys, i = h*64+dh)
//   Vt[((t*NHEAD+h)*DH + dh)*LPAD + c]        (dh-major values)
__global__ void k_build_kv(const _Float16* __restrict__ KVx,
                           const _Float16* __restrict__ KVl,
                           _Float16* __restrict__ Kc,
                           _Float16* __restrict__ Vt) {
  size_t idx = (size_t)blockIdx.x * 256 + threadIdx.x;
  const size_t total = (size_t)TT * LPAD * 128;
  if (idx >= total) return;
  int i    = (int)(idx % 128);                // h*64 + dh
  int rest = (int)(idx / 128);
  int c = rest % LPAD;
  int t = rest / LPAD;
  _Float16 kv_ = (_Float16)0.f, vv_ = (_Float16)0.f;
  if (c < LTOT) {
    const _Float16* rp = (c < SS) ? (KVx + ((size_t)t * SS + c) * 256)
                                  : (KVl + ((size_t)t * NLAT + (c - SS)) * 256);
    kv_ = rp[i];
    vv_ = rp[128 + i];
  }
  Kc[idx] = kv_;
  int h = i >> 6, dh = i & 63;
  Vt[((size_t)(t * NHEAD + h) * DH + dh) * LPAD + c] = vv_;
}

// ---------------- flash attention, one wave per (t, head, 16-latent-row tile) ----------
// Q:(T,NLAT,128) f16. Kc/Vt zero-padded per above. Out:(T,NLAT,128) f16.
__global__ void k_attn(const _Float16* __restrict__ Q,
                       const _Float16* __restrict__ Kc,
                       const _Float16* __restrict__ Vt,
                       const int* __restrict__ split,
                       _Float16* __restrict__ Out) {
  __shared__ _Float16 P[16][32] __attribute__((aligned(64)));
  const int lane = threadIdx.x & 31;
  const int l15  = lane & 15;
  const int hi   = lane >> 4;
  const int tile = blockIdx.x;
  const int h    = blockIdx.y;
  const int t    = blockIdx.z;
  const int valid = split[t] * VV;            // unmasked visual columns
  const int kbase = hi * 16;
  const int kn    = hi * 16;

  // Q A-fragments for K(dh)=0..31 and 32..63
  v16h q0 = {}, q1 = {};
  {
    int row = tile * 16 + l15;
    if (row >= NLAT) row = NLAT - 1;          // clamp: those C rows are never stored
    const _Float16* qp = Q + ((size_t)(t * NLAT + row)) * INNER + h * DH;
    q0 = cat8(*(const v8h*)(qp + kbase),      *(const v8h*)(qp + 8 + kbase));
    q1 = cat8(*(const v8h*)(qp + 32 + kbase), *(const v8h*)(qp + 40 + kbase));
  }

  const _Float16* kcb = Kc + (size_t)t * LPAD * 128 + h * DH + kn;
  const _Float16* vtb = Vt + ((size_t)(t * NHEAD + h) * DH) * LPAD;

  float m[8], l[8];
  v8f o[4] = {};
#pragma unroll
  for (int r = 0; r < 8; ++r) { m[r] = -1e30f; l[r] = 0.f; }

  for (int c0 = 0; c0 < LPAD; c0 += 32) {
    // ---- scores: q(16x64) @ K^T for two 16-col tiles (unconditional b128 loads) ----
    v8f s[2];
#pragma unroll
    for (int nt = 0; nt < 2; ++nt) {
      const _Float16* kp = kcb + (size_t)(c0 + nt * 16 + l15) * 128;
      v16h b0 = *(const v16h*)kp;
      v16h b1 = *(const v16h*)(kp + 32);
      v8f sc = {};
      sc = __builtin_amdgcn_wmma_f32_16x16x32_f16(false, q0, false, b0, (short)0, sc, false, false);
      sc = __builtin_amdgcn_wmma_f32_16x16x32_f16(false, q1, false, b1, (short)0, sc, false, false);
      s[nt] = sc;
    }
    // ---- scale + mask (column is per-lane constant in C layout) ----
#pragma unroll
    for (int nt = 0; nt < 2; ++nt) {
      int colc = c0 + nt * 16 + l15;
      bool msk = (colc < SS) ? (colc >= valid) : (colc >= LTOT);
#pragma unroll
      for (int r = 0; r < 8; ++r) {
        float v = s[nt][r] * 0.125f;          // dh^-0.5
        s[nt][r] = msk ? -1e30f : v;
      }
    }
    // ---- online softmax (rows live across the 16 lanes of each half) ----
    float mnew[8], corr[8];
#pragma unroll
    for (int r = 0; r < 8; ++r) {
      float v = fmaxf(s[0][r], s[1][r]);
#pragma unroll
      for (int off = 8; off >= 1; off >>= 1) v = fmaxf(v, __shfl_xor(v, off, 32));
      mnew[r] = fmaxf(m[r], v);
      corr[r] = __expf(m[r] - mnew[r]);
      m[r] = mnew[r];
    }
#pragma unroll
    for (int r = 0; r < 8; ++r) {
      float p0 = (s[0][r] <= -1e29f) ? 0.f : __expf(s[0][r] - mnew[r]);
      float p1 = (s[1][r] <= -1e29f) ? 0.f : __expf(s[1][r] - mnew[r]);
      float rs = p0 + p1;
#pragma unroll
      for (int off = 8; off >= 1; off >>= 1) rs += __shfl_xor(rs, off, 32);
      l[r] = l[r] * corr[r] + rs;
      int prow = r + hi * 8;
      P[prow][l15]      = (_Float16)p0;
      P[prow][16 + l15] = (_Float16)p1;
#pragma unroll
      for (int a2 = 0; a2 < 4; ++a2) o[a2][r] *= corr[r];
    }
    __syncthreads();
    // ---- P (16x32) as A-fragment via LDS transpose ----
    v16h pf;
    {
      const _Float16* pp = &P[l15][0];
      pf = cat8(*(const v8h*)(pp + kbase), *(const v8h*)(pp + 8 + kbase));
    }
    // ---- O += P @ V over 4 dh tiles (contiguous Vt fragments) ----
#pragma unroll
    for (int a2 = 0; a2 < 4; ++a2) {
      int dh = a2 * 16 + l15;
      v16h vf = *(const v16h*)(vtb + (size_t)dh * LPAD + c0 + kn);
      o[a2] = __builtin_amdgcn_wmma_f32_16x16x32_f16(false, pf, false, vf,
                                                     (short)0, o[a2], false, false);
    }
    __syncthreads();
  }
  // ---- normalize + store ----
#pragma unroll
  for (int a2 = 0; a2 < 4; ++a2) {
#pragma unroll
    for (int r = 0; r < 8; ++r) {
      int row = tile * 16 + r + hi * 8;
      if (row < NLAT) {
        float v = o[a2][r] / l[r];
        Out[((size_t)(t * NLAT + row)) * INNER + h * DH + a2 * 16 + l15] = (_Float16)v;
      }
    }
  }
}

// ---------------- host orchestration ----------------
static inline char* bump(char*& p, size_t bytes) {
  char* r = p;
  p += (bytes + 255) & ~(size_t)255;
  return r;
}

extern "C" void kernel_launch(void* const* d_in, const int* in_sizes, int n_in,
                              void* d_out, int out_size, void* d_ws, size_t ws_size,
                              hipStream_t stream) {
  (void)in_sizes; (void)out_size; (void)ws_size;
  if (n_in < 30) return;

  const float* x       = (const float*)d_in[0];
  const int*   split   = (const int*)d_in[1];
  const float* img_cls = (const float*)d_in[2];
  const float* p_lat   = (const float*)d_in[3];
  const float* p_sp    = (const float*)d_in[4];
  const float* p_fr    = (const float*)d_in[5];
  const float* fin_g   = (const float*)d_in[6 + 11 * 2 + 0];
  const float* fin_b   = (const float*)d_in[6 + 11 * 2 + 1];

  char* wp = (char*)d_ws;
  _Float16* xn    = (_Float16*)bump(wp, (size_t)TT * SS * DD * 2);
  _Float16* lnl   = (_Float16*)bump(wp, (size_t)TT * NLAT * DD * 2);
  _Float16* kvx   = (_Float16*)bump(wp, (size_t)TT * SS * 256 * 2);
  _Float16* kvl   = (_Float16*)bump(wp, (size_t)TT * NLAT * 256 * 2);
  _Float16* kc    = (_Float16*)bump(wp, (size_t)TT * LPAD * 128 * 2);
  _Float16* vt    = (_Float16*)bump(wp, (size_t)TT * NHEAD * DH * LPAD * 2);
  _Float16* qbuf  = (_Float16*)bump(wp, (size_t)TT * NLAT * INNER * 2);
  _Float16* abuf  = (_Float16*)bump(wp, (size_t)TT * NLAT * INNER * 2);
  float*    lat   = (float*)bump(wp, (size_t)TT * NLAT * DD * 4);
  _Float16* hff   = (_Float16*)bump(wp, (size_t)TT * NLAT * FFDIM * 2);
  _Float16* wqt   = (_Float16*)bump(wp, (size_t)DD * INNER * 2);   // INNER x DD
  _Float16* wkvt  = (_Float16*)bump(wp, (size_t)DD * 256 * 2);     // 256 x DD
  _Float16* woutt = (_Float16*)bump(wp, (size_t)INNER * DD * 2);   // DD x INNER
  _Float16* w1t   = (_Float16*)bump(wp, (size_t)DD * FFDIM * 2);   // FFDIM x DD
  _Float16* w2t   = (_Float16*)bump(wp, (size_t)FFDIM * DD * 2);   // DD x FFDIM

  const int Mlat = TT * NLAT;                 // 2884 latent rows total
  auto cvt_t = [&](const float* s, _Float16* d, int K, int N) {
    size_t n = (size_t)K * N;
    k_cvt_t<<<(unsigned)((n + 255) / 256), 256, 0, stream>>>(s, d, K, N);
  };

  k_init_latents<<<(unsigned)(((size_t)TT * NLAT * DD + 255) / 256), 256, 0, stream>>>(
      p_lat, img_cls, lat);

  for (int layer = 0; layer < 2; ++layer) {
    const float* nm_g = (const float*)d_in[6 + 11 * layer + 0];
    const float* nm_b = (const float*)d_in[6 + 11 * layer + 1];
    const float* nl_g = (const float*)d_in[6 + 11 * layer + 2];
    const float* nl_b = (const float*)d_in[6 + 11 * layer + 3];
    const float* Wq   = (const float*)d_in[6 + 11 * layer + 4];
    const float* Wkv  = (const float*)d_in[6 + 11 * layer + 5];
    const float* Wout = (const float*)d_in[6 + 11 * layer + 6];
    const float* ff_g = (const float*)d_in[6 + 11 * layer + 7];
    const float* ff_b = (const float*)d_in[6 + 11 * layer + 8];
    const float* W1   = (const float*)d_in[6 + 11 * layer + 9];
    const float* W2   = (const float*)d_in[6 + 11 * layer + 10];

    cvt_t(Wq,   wqt,   DD,    INNER);
    cvt_t(Wkv,  wkvt,  DD,    256);
    cvt_t(Wout, woutt, INNER, DD);
    cvt_t(W1,   w1t,   DD,    FFDIM);
    cvt_t(W2,   w2t,   FFDIM, DD);

    // xn = LN(x + embeds); KVx = xn @ Wkv
    k_ln_x_embed<<<TT * SS, 256, 0, stream>>>(x, p_sp, p_fr, nm_g, nm_b, xn);
    k_gemm<<<dim3(256 / 32, (TT * SS) / 16), 32, 0, stream>>>(
        xn, wkvt, nullptr, kvx, TT * SS, 256, DD, 1);

    // latent-side LN, Q, KV
    k_ln_rows<<<Mlat, 256, 0, stream>>>(lat, nl_g, nl_b, lnl, 1);
    k_gemm<<<dim3(INNER / 32, (Mlat + 15) / 16), 32, 0, stream>>>(
        lnl, wqt, nullptr, qbuf, Mlat, INNER, DD, 1);
    k_gemm<<<dim3(256 / 32, (Mlat + 15) / 16), 32, 0, stream>>>(
        lnl, wkvt, nullptr, kvl, Mlat, 256, DD, 1);

    // repack into zero-padded attention layouts (token-major K, dh-major V)
    k_build_kv<<<(unsigned)(((size_t)TT * LPAD * 128 + 255) / 256), 256, 0, stream>>>(
        kvx, kvl, kc, vt);

    // flash attention over S + NLAT columns
    k_attn<<<dim3((NLAT + 15) / 16, NHEAD, TT), 32, 0, stream>>>(
        qbuf, kc, vt, split, abuf);

    // out proj + residual
    k_gemm<<<dim3(DD / 32, (Mlat + 15) / 16), 32, 0, stream>>>(
        abuf, woutt, lat, lat, Mlat, DD, INNER, 0);

    // feed-forward: LN -> W1 -> GELU -> W2 + residual
    k_ln_rows<<<Mlat, 256, 0, stream>>>(lat, ff_g, ff_b, lnl, 1);
    k_gemm<<<dim3(FFDIM / 32, (Mlat + 15) / 16), 32, 0, stream>>>(
        lnl, w1t, nullptr, hff, Mlat, FFDIM, DD, 1 | 2);
    k_gemm<<<dim3(DD / 32, (Mlat + 15) / 16), 32, 0, stream>>>(
        hff, w2t, lat, lat, Mlat, DD, FFDIM, 0);
  }

  // final LayerNorm -> f32 output
  k_ln_rows<<<Mlat, 256, 0, stream>>>(lat, fin_g, fin_b, d_out, 0);
}
